// DocEncoder_2516850836026
// MI455X (gfx1250) — matrix operand order repaired and checked by reading, not compiled, not run on hardware
//
#include <hip/hip_runtime.h>
#include <hip/hip_bf16.h>
#include <math.h>

// ---- problem dims ----
#define BB   4096
#define SS   64
#define DD   300
#define HH   6
#define EE   400
#define VSS  200
#define DHH  50
#define DP   320      // D padded to 10 k-chunks of 32
#define KC_D 10       // k-chunks for K=300->320
#define NT_D 19       // n-tiles for N=300->304
#define MT_D 4        // m-tiles for 64 rows
#define EP   416      // E padded for K of Wa GEMM (400 -> 13*32)
#define KC_E 13
#define NT_E 25       // n-tiles for N=400
#define NT_V 13       // n-tiles for N=200->208

typedef __attribute__((ext_vector_type(16))) __bf16 v16bf;
typedef __attribute__((ext_vector_type(8)))  __bf16 v8bf;
typedef __attribute__((ext_vector_type(8)))  float  v8f;

// ---- packed-weight sizes (bf16 elements) ----
#define PK_QKV (NT_D * KC_D * 512)      // 97280  (also Wo)
#define PK_P   (NT_E * KC_D * 512)      // 128000
#define PK_A   (NT_V * KC_E * 512)      // 86528
#define PK_R   (4 * 2 * 512)            // 4096

// ---- LDS layout (bytes) ----
#define OFF_XB 0        // bf16[64][320]  x   -> later r
#define OFF_QB 40960    // bf16[64][320]  q   -> later attn_out
#define OFF_KB 81920    // bf16[64][320]  k/p
#define OFF_HB 81920    // bf16[64][416]  h (bf16)   [overlaps dead k + head of v]
#define OFF_VB 122880   // bf16[64][320]  v/u
#define OFF_HF 135168   // f32 [64][400]  h (f32)    [overlaps dead tail of v]
#define OFF_SA 237568   // f32 [6][64]    alpha scores / alpha
#define OFF_G  239104   // f32 [6][64]    global query g
#define OFF_SB 240640   // f32 [6][64]    beta scores / beta
#define OFF_GK 242176   // f32 [6][64]    global key gk
#define OFF_SC 243712   // f32 [64]       pooling scores
#define OFF_WM 243968   // f32 [64]       pooling softmax weights
#define LDS_BYTES 244224

// ---------------------------------------------------------------------------
// Weight repack: W[K][N] (f32, row-major) -> per-fragment bf16 layout so a
// B-fragment (32x16, 16-bit) is one contiguous 32B read per lane.
//   out[(((nt*KC)+kc)*32 + lane)*16 + j] = W[kc*32 + (lane<16?0:8) + (j&7) + (j<8?0:16)][nt*16 + (lane&15)]
// ---------------------------------------------------------------------------
__global__ void __launch_bounds__(256)
repack_kernel(const float* __restrict__ W, int K, int N, int KC, int NT,
              __bf16* __restrict__ out) {
    int idx = blockIdx.x * 256 + threadIdx.x;
    int total = NT * KC * 512;
    if (idx >= total) return;
    int j    = idx & 15;
    int lane = (idx >> 4) & 31;
    int rest = idx >> 9;
    int kc   = rest % KC;
    int nt   = rest / KC;
    int n = nt * 16 + (lane & 15);
    int k = kc * 32 + ((lane >> 4) << 3) + (j & 7) + ((j >> 3) << 4);
    float v = (k < K && n < N) ? W[(size_t)k * N + n] : 0.0f;
    out[idx] = (__bf16)v;
}

// ---- fragment loaders ----
// A (16x32 bf16) from an LDS row-major matrix with leading dim `ld`.
// lanes 0-15: M=lane, K = {0..7, 16..23}; lanes 16-31: M=lane-16, K = {8..15, 24..31}
__device__ __forceinline__ v16bf frag_a(const __bf16* base, int ld, int mt, int kc, int lane) {
    int row  = mt * 16 + (lane & 15);
    int koff = kc * 32 + ((lane & 16) ? 8 : 0);
    const __bf16* p = base + row * ld + koff;
    v8bf lo = *(const v8bf*)(p);
    v8bf hi = *(const v8bf*)(p + 16);
    v16bf r;
#pragma unroll
    for (int j = 0; j < 8; ++j) { r[j] = lo[j]; r[j + 8] = hi[j]; }
    return r;
}

// B (32x16 bf16) from packed global weights: contiguous 32B per lane.
__device__ __forceinline__ v16bf frag_b(const __bf16* pack, int KC, int nt, int kc, int lane) {
    const __bf16* p = pack + (((size_t)nt * KC + kc) * 32 + lane) * 16;
    v8bf lo = *(const v8bf*)(p);
    v8bf hi = *(const v8bf*)(p + 8);
    v16bf r;
#pragma unroll
    for (int j = 0; j < 8; ++j) { r[j] = lo[j]; r[j + 8] = hi[j]; }
    return r;
}

#define WMMA_BF16(A, Bf, C) \
    __builtin_amdgcn_wmma_f32_16x16x32_bf16(false, (A), false, (Bf), (short)0, (C), false, false)

// ---------------------------------------------------------------------------
// Fully fused Fastformer document encoder: one block per batch element.
// ---------------------------------------------------------------------------
__global__ void __launch_bounds__(256)
fused_doc_encoder(const int*   __restrict__ tokens,
                  const float* __restrict__ emb,
                  const float* __restrict__ wq_attn,
                  const float* __restrict__ wk_attn,
                  const float* __restrict__ br,
                  const float* __restrict__ bo,
                  const float* __restrict__ bp,
                  const float* __restrict__ ba,
                  const float* __restrict__ va,
                  const __bf16* __restrict__ pWq,
                  const __bf16* __restrict__ pWk,
                  const __bf16* __restrict__ pWv,
                  const __bf16* __restrict__ pWr,
                  const __bf16* __restrict__ pWo,
                  const __bf16* __restrict__ pWp,
                  const __bf16* __restrict__ pWa,
                  float* __restrict__ out) {
    extern __shared__ char smem[];
    __bf16* xb = (__bf16*)(smem + OFF_XB);   // x, later r
    __bf16* qb = (__bf16*)(smem + OFF_QB);   // q, later attn_out
    __bf16* kb = (__bf16*)(smem + OFF_KB);   // k, then p
    __bf16* vb = (__bf16*)(smem + OFF_VB);   // v, then u
    __bf16* hb = (__bf16*)(smem + OFF_HB);   // h (bf16, K-padded)
    float*  hf = (float*) (smem + OFF_HF);   // h (f32)
    float*  sa = (float*) (smem + OFF_SA);   // alpha
    float*  gq = (float*) (smem + OFF_G);    // g
    float*  sb = (float*) (smem + OFF_SB);   // beta
    float*  gk = (float*) (smem + OFF_GK);   // gk
    float*  sc = (float*) (smem + OFF_SC);   // pooling scores
    float*  wm = (float*) (smem + OFF_WM);   // pooling weights

    const int b    = blockIdx.x;
    const int tid  = threadIdx.x;
    const int w    = tid >> 5;
    const int lane = tid & 31;
    const float scale = 0.14142135623730951f;  // DH^-0.5

    // ---- pass 0: embedding gather -> bf16 x, zero K-pad; clear score accum ----
    for (int i = tid; i < SS * DP; i += 256) {
        int s = i / DP, c = i - s * DP;
        float v = 0.0f;
        if (c < DD) v = emb[(size_t)tokens[b * SS + s] * DD + c];
        xb[s * DP + c] = (__bf16)v;
    }
    if (tid < SS) sc[tid] = 0.0f;
    __syncthreads();

    // ---- pass 1: Q,K,V = x @ {Wq,Wk,Wv}  (shared A fragment, 3 WMMAs) ----
    for (int t = w; t < MT_D * NT_D; t += 8) {
        int mt = t / NT_D, nt = t - mt * NT_D;
        v8f cq = {}; v8f ck = {}; v8f cv = {};
        for (int kc = 0; kc < KC_D; ++kc) {
            v16bf a  = frag_a(xb, DP, mt, kc, lane);
            v16bf bq = frag_b(pWq, KC_D, nt, kc, lane);
            v16bf bk = frag_b(pWk, KC_D, nt, kc, lane);
            v16bf bv = frag_b(pWv, KC_D, nt, kc, lane);
            cq = WMMA_BF16(a, bq, cq);
            ck = WMMA_BF16(a, bk, ck);
            cv = WMMA_BF16(a, bv, cv);
        }
        int n = nt * 16 + (lane & 15);
        if (n < DD) {
            int rbase = mt * 16 + ((lane & 16) ? 8 : 0);
#pragma unroll
            for (int r = 0; r < 8; ++r) {
                int row = rbase + r;
                qb[row * DP + n] = (__bf16)cq[r];
                kb[row * DP + n] = (__bf16)ck[r];
                vb[row * DP + n] = (__bf16)cv[r];
            }
        }
    }
    __syncthreads();

    // ---- pass 2: alpha scores  sa[h][s] = scale * sum_d q[s,h,d]*wq_attn[h,d] ----
    for (int i = tid; i < HH * SS; i += 256) {
        int h = i >> 6, s = i & 63;
        float acc = 0.0f;
        for (int d = 0; d < DHH; ++d)
            acc += (float)qb[s * DP + h * DHH + d] * wq_attn[h * DHH + d];
        sa[h * 64 + s] = acc * scale;
    }
    __syncthreads();

    // ---- pass 3: softmax over s for each head (alpha) ----
    if (tid < HH) {
        float mx = -1e30f;
        for (int s = 0; s < SS; ++s) mx = fmaxf(mx, sa[tid * 64 + s]);
        float sum = 0.0f;
        for (int s = 0; s < SS; ++s) { float e = expf(sa[tid * 64 + s] - mx); sa[tid * 64 + s] = e; sum += e; }
        float inv = 1.0f / sum;
        for (int s = 0; s < SS; ++s) sa[tid * 64 + s] *= inv;
    }
    __syncthreads();

    // ---- pass 4: global query g[h][d] = sum_s alpha[h][s]*q[s,h,d] ----
    for (int i = tid; i < HH * DHH; i += 256) {
        int h = i / DHH, d = i - h * DHH;
        float acc = 0.0f;
        for (int s = 0; s < SS; ++s)
            acc += sa[h * 64 + s] * (float)qb[s * DP + h * DHH + d];
        gq[h * 64 + d] = acc;
    }
    __syncthreads();

    // ---- pass 5: p = g (broadcast) * k, in place ----
    for (int i = tid; i < SS * DD; i += 256) {
        int s = i / DD, c = i - s * DD;
        int h = c / DHH, d = c - h * DHH;
        kb[s * DP + c] = (__bf16)(gq[h * 64 + d] * (float)kb[s * DP + c]);
    }
    __syncthreads();

    // ---- pass 6: beta scores from p ----
    for (int i = tid; i < HH * SS; i += 256) {
        int h = i >> 6, s = i & 63;
        float acc = 0.0f;
        for (int d = 0; d < DHH; ++d)
            acc += (float)kb[s * DP + h * DHH + d] * wk_attn[h * DHH + d];
        sb[h * 64 + s] = acc * scale;
    }
    __syncthreads();

    // ---- pass 7: softmax (beta) ----
    if (tid < HH) {
        float mx = -1e30f;
        for (int s = 0; s < SS; ++s) mx = fmaxf(mx, sb[tid * 64 + s]);
        float sum = 0.0f;
        for (int s = 0; s < SS; ++s) { float e = expf(sb[tid * 64 + s] - mx); sb[tid * 64 + s] = e; sum += e; }
        float inv = 1.0f / sum;
        for (int s = 0; s < SS; ++s) sb[tid * 64 + s] *= inv;
    }
    __syncthreads();

    // ---- pass 8: global key gk[h][d] = sum_s beta[h][s]*p[s,h,d] ----
    for (int i = tid; i < HH * DHH; i += 256) {
        int h = i / DHH, d = i - h * DHH;
        float acc = 0.0f;
        for (int s = 0; s < SS; ++s)
            acc += sb[h * 64 + s] * (float)kb[s * DP + h * DHH + d];
        gk[h * 64 + d] = acc;
    }
    __syncthreads();

    // ---- pass 9: u = gk (broadcast) * v in place; zero K-pad of r buffer ----
    for (int i = tid; i < SS * DD; i += 256) {
        int s = i / DD, c = i - s * DD;
        int h = c / DHH, d = c - h * DHH;
        vb[s * DP + c] = (__bf16)(gk[h * 64 + d] * (float)vb[s * DP + c]);
    }
    for (int i = tid; i < SS * (DP - DD); i += 256) {
        int s = i / (DP - DD), j = i - s * (DP - DD);
        xb[s * DP + DD + j] = (__bf16)0.0f;     // xb now becomes r
    }
    __syncthreads();

    // ---- pass 10: r = u @ Wr + br + q  (heads stacked along M: 384x50x50) ----
    for (int t = w; t < 24 * 4; t += 8) {
        int mt = t >> 2, nt = t & 3;
        int m = mt * 16 + (lane & 15);
        int hh2 = m >> 6, s2 = m & 63;
        int kbase = (lane & 16) ? 8 : 0;
        v8f c = {};
        for (int kc = 0; kc < 2; ++kc) {
            v16bf a;
#pragma unroll
            for (int j = 0; j < 16; ++j) {
                int d = kc * 32 + kbase + j + ((j >= 8) ? 8 : 0);
                float uv = (d < DHH) ? (float)vb[s2 * DP + hh2 * DHH + d] : 0.0f;
                a[j] = (__bf16)uv;
            }
            v16bf bw = frag_b(pWr, 2, nt, kc, lane);
            c = WMMA_BF16(a, bw, c);
        }
        int n = nt * 16 + (lane & 15);
        if (n < DHH) {
            int rbase = mt * 16 + ((lane & 16) ? 8 : 0);
#pragma unroll
            for (int r = 0; r < 8; ++r) {
                int m2 = rbase + r;
                int h2 = m2 >> 6, s3 = m2 & 63;
                int col = h2 * DHH + n;
                xb[s3 * DP + col] = (__bf16)(c[r] + br[n] + (float)qb[s3 * DP + col]);
            }
        }
    }
    __syncthreads();

    // ---- pass 11: attn_out = r @ Wo + bo  -> qb region (bf16); zero its K-pad ----
    for (int i = tid; i < SS * (DP - DD); i += 256) {
        int s = i / (DP - DD), j = i - s * (DP - DD);
        qb[s * DP + DD + j] = (__bf16)0.0f;
    }
    for (int t = w; t < MT_D * NT_D; t += 8) {
        int mt = t / NT_D, nt = t - mt * NT_D;
        v8f c = {};
        for (int kc = 0; kc < KC_D; ++kc) {
            v16bf a  = frag_a(xb, DP, mt, kc, lane);
            v16bf bw = frag_b(pWo, KC_D, nt, kc, lane);
            c = WMMA_BF16(a, bw, c);
        }
        int n = nt * 16 + (lane & 15);
        if (n < DD) {
            int rbase = mt * 16 + ((lane & 16) ? 8 : 0);
#pragma unroll
            for (int r = 0; r < 8; ++r)
                qb[(rbase + r) * DP + n] = (__bf16)(c[r] + bo[n]);
        }
    }
    __syncthreads();

    // ---- pass 12: h = attn_out @ Wp + bp  -> hf (f32) and hb (bf16, K-padded) ----
    for (int i = tid; i < SS * (EP - EE); i += 256) {
        int s = i / (EP - EE), j = i - s * (EP - EE);
        hb[s * EP + EE + j] = (__bf16)0.0f;
    }
    for (int t = w; t < MT_D * NT_E; t += 8) {
        int mt = t / NT_E, nt = t - mt * NT_E;
        v8f c = {};
        for (int kc = 0; kc < KC_D; ++kc) {
            v16bf a  = frag_a(qb, DP, mt, kc, lane);
            v16bf bw = frag_b(pWp, KC_D, nt, kc, lane);
            c = WMMA_BF16(a, bw, c);
        }
        int n = nt * 16 + (lane & 15);
        int rbase = mt * 16 + ((lane & 16) ? 8 : 0);
#pragma unroll
        for (int r = 0; r < 8; ++r) {
            float hv = c[r] + bp[n];
            hf[(rbase + r) * EE + n] = hv;
            hb[(rbase + r) * EP + n] = (__bf16)hv;
        }
    }
    __syncthreads();

    // ---- pass 13: scores[s] += tanh(h @ Wa + ba)[s,n] * va[n] ----
    for (int t = w; t < MT_D * NT_V; t += 8) {
        int mt = t / NT_V, nt = t - mt * NT_V;
        v8f c = {};
        for (int kc = 0; kc < KC_E; ++kc) {
            v16bf a  = frag_a(hb, EP, mt, kc, lane);
            v16bf bw = frag_b(pWa, KC_E, nt, kc, lane);
            c = WMMA_BF16(a, bw, c);
        }
        int n = nt * 16 + (lane & 15);
        if (n < VSS) {
            float van = va[n];
            int rbase = mt * 16 + ((lane & 16) ? 8 : 0);
#pragma unroll
            for (int r = 0; r < 8; ++r) {
                float tv = tanhf(c[r] + ba[n]);
                atomicAdd(&sc[rbase + r], tv * van);
            }
        }
    }
    __syncthreads();

    // ---- pass 14: softmax over sequence -> pooling weights ----
    if (tid == 0) {
        float mx = -1e30f;
        for (int s = 0; s < SS; ++s) mx = fmaxf(mx, sc[s]);
        float sum = 0.0f;
        for (int s = 0; s < SS; ++s) { float e = expf(sc[s] - mx); wm[s] = e; sum += e; }
        float inv = 1.0f / sum;
        for (int s = 0; s < SS; ++s) wm[s] *= inv;
    }
    __syncthreads();

    // ---- pass 15: pooled[e] = sum_s wm[s] * h[s][e] ----
    for (int e = tid; e < EE; e += 256) {
        float acc = 0.0f;
        for (int s = 0; s < SS; ++s) acc += wm[s] * hf[s * EE + e];
        out[(size_t)b * EE + e] = acc;
    }
}

// ---------------------------------------------------------------------------
extern "C" void kernel_launch(void* const* d_in, const int* in_sizes, int n_in,
                              void* d_out, int out_size, void* d_ws, size_t ws_size,
                              hipStream_t stream) {
    const int*   tokens  = (const int*)  d_in[0];
    const float* emb     = (const float*)d_in[1];
    const float* Wq      = (const float*)d_in[2];
    const float* Wk      = (const float*)d_in[3];
    const float* Wv      = (const float*)d_in[4];
    const float* wq_attn = (const float*)d_in[5];
    const float* wk_attn = (const float*)d_in[6];
    const float* Wr      = (const float*)d_in[7];
    const float* br      = (const float*)d_in[8];
    const float* Wo      = (const float*)d_in[9];
    const float* bo      = (const float*)d_in[10];
    const float* Wp      = (const float*)d_in[11];
    const float* bp      = (const float*)d_in[12];
    const float* Wa      = (const float*)d_in[13];
    const float* ba      = (const float*)d_in[14];
    const float* va      = (const float*)d_in[15];

    __bf16* ws  = (__bf16*)d_ws;
    __bf16* pWq = ws;
    __bf16* pWk = pWq + PK_QKV;
    __bf16* pWv = pWk + PK_QKV;
    __bf16* pWo = pWv + PK_QKV;
    __bf16* pWp = pWo + PK_QKV;
    __bf16* pWa = pWp + PK_P;
    __bf16* pWr = pWa + PK_A;

    auto repack = [&](const float* W, int K, int N, int KC, int NT, __bf16* outp) {
        int total = NT * KC * 512;
        repack_kernel<<<dim3((total + 255) / 256), dim3(256), 0, stream>>>(W, K, N, KC, NT, outp);
    };
    repack(Wq, DD, DD, KC_D, NT_D, pWq);
    repack(Wk, DD, DD, KC_D, NT_D, pWk);
    repack(Wv, DD, DD, KC_D, NT_D, pWv);
    repack(Wo, DD, DD, KC_D, NT_D, pWo);
    repack(Wp, DD, EE, KC_D, NT_E, pWp);
    repack(Wa, EE, VSS, KC_E, NT_V, pWa);
    repack(Wr, DHH, DHH, 2, 4, pWr);

    (void)hipFuncSetAttribute((const void*)fused_doc_encoder,
                              hipFuncAttributeMaxDynamicSharedMemorySize, LDS_BYTES);

    fused_doc_encoder<<<dim3(BB), dim3(256), LDS_BYTES, stream>>>(
        tokens, emb, wq_attn, wk_attn, br, bo, bp, ba, va,
        pWq, pWk, pWv, pWr, pWo, pWp, pWa, (float*)d_out);
}
